// EncoderLayer_78331613544668
// MI455X (gfx1250) — matrix operand order repaired and checked by reading, not compiled, not run on hardware
//
#include <hip/hip_runtime.h>
#include <hip/hip_bf16.h>

typedef __attribute__((ext_vector_type(16))) _Float16 v16h;
typedef __attribute__((ext_vector_type(8)))  _Float16 v8h;
typedef __attribute__((ext_vector_type(8)))  float    v8f;
typedef __attribute__((ext_vector_type(4)))  unsigned int u32x4;
typedef __attribute__((ext_vector_type(8)))  int i32x8;
typedef __attribute__((ext_vector_type(4)))  int i32x4;

#define BATCH 2
#define SEQ   4096
#define DMODEL 512
#define NHEADS 8
#define DHEAD 64
#define DFF   2048
#define MROWS (BATCH*SEQ)   // 8192
#define LNEPS 1e-5f
#define KCHUNK 128          // K-elements staged per TDM transfer (4 WMMA K-steps)

// ---------------------------------------------------------------------------
// TDM: issue a 2D tile load (rows x kc f16 elements, row stride strideElems)
// from global Bt into LDS at byte offset lds_off. D# packed per ISA 8.3/8.4.
// 6-arg builtin variant (clang-23 / therock headers).
// ---------------------------------------------------------------------------
__device__ __forceinline__
void tdm_load_tile(const _Float16* gptr, unsigned lds_off,
                   int rows, int kc, int strideElems) {
    unsigned long long ga = (unsigned long long)(const void*)gptr;
    // group0: count=1 | lds_addr | global_addr[56:0] | type=2
    u32x4 g0 = { 1u,
                 lds_off,
                 (unsigned)ga,
                 (unsigned)((ga >> 32) & 0x1FFFFFFull) | (2u << 30) };
    // group1: data_size=2B; tensor_dim0=kc; tensor_dim1=rows;
    //         tile_dim0=kc; tile_dim1=rows; tile_dim2=0;
    //         tensor_dim0_stride=strideElems; tensor_dim1_stride=0
    i32x8 g1 = { (int)(1u << 16),             // d0: data_size=1 (2 bytes)
                 (int)((unsigned)kc << 16),   // d1: tensor_dim0[15:0] << 16
                 (int)((unsigned)rows << 16), // d2: td0_hi(0) | tensor_dim1[15:0]<<16
                 (int)((unsigned)kc << 16),   // d3: td1_hi(0) | tile_dim0<<16
                 rows,                        // d4: tile_dim1 | tile_dim2(0)<<16
                 strideElems,                 // d5: tensor_dim0_stride[31:0]
                 0,                           // d6: stride hi | td1_stride lo
                 0 };                         // d7
    i32x4 z4 = { 0, 0, 0, 0 };
    i32x8 z8 = { 0, 0, 0, 0, 0, 0, 0, 0 };
    __builtin_amdgcn_tensor_load_to_lds(g0, g1, z4, z4, z8, 0);
}

// ---------------------------------------------------------------------------
// fp32 [K,N] weight -> f16 transposed [N,K]
// ---------------------------------------------------------------------------
__global__ void convT_kernel(const float* __restrict__ W, _Float16* __restrict__ Wt,
                             int K, int N) {
    int idx = blockIdx.x * 256 + threadIdx.x;
    if (idx >= K * N) return;
    int k = idx / N, n = idx - k * N;
    Wt[(long)n * K + k] = (_Float16)W[idx];
}

// ---------------------------------------------------------------------------
// LayerNorm: one wave per row of 512, f16 output
// ---------------------------------------------------------------------------
__global__ __launch_bounds__(256)
void ln_kernel(const float* __restrict__ x, const float* __restrict__ g,
               const float* __restrict__ b, _Float16* __restrict__ out) {
    int wid  = threadIdx.x >> 5;
    int lane = threadIdx.x & 31;
    int row  = blockIdx.x * 8 + wid;
    const float* xr = x + (long)row * DMODEL + lane * 16;
    float v[16];
    const float4* x4 = (const float4*)xr;
    #pragma unroll
    for (int i = 0; i < 4; i++) {
        float4 t = x4[i];
        v[i*4+0]=t.x; v[i*4+1]=t.y; v[i*4+2]=t.z; v[i*4+3]=t.w;
    }
    float s = 0.f, q = 0.f;
    #pragma unroll
    for (int i = 0; i < 16; i++) { s += v[i]; q += v[i]*v[i]; }
    #pragma unroll
    for (int off = 16; off >= 1; off >>= 1) {
        s += __shfl_xor(s, off, 32);
        q += __shfl_xor(q, off, 32);
    }
    float mu = s * (1.0f / DMODEL);
    float var = q * (1.0f / DMODEL) - mu * mu;
    float rs = rsqrtf(var + LNEPS);
    _Float16* o = out + (long)row * DMODEL + lane * 16;
    #pragma unroll
    for (int i = 0; i < 16; i++) {
        int c = lane * 16 + i;
        o[i] = (_Float16)((v[i] - mu) * rs * g[c] + b[c]);
    }
}

// ---------------------------------------------------------------------------
// WMMA GEMM with TDM-staged, double-buffered B tile in LDS, ping-pong
// B-fragment registers, and sched_group_barrier-pinned software pipeline.
// Block = 8 waves sharing tileN (64 cols); wave w -> tileM = base+w.
// mode 0: f16 row-major     mode 1: f16 row-major + ReLU
// mode 2: f16 scatter [B,H,S,Dh]   mode 3: f16 scatter [B,H,Dh,S]
// mode 4: f32 row-major out = acc + bias + residual
// Requires K % KCHUNK == 0, KCHUNK == 128.
// ---------------------------------------------------------------------------
__global__ __launch_bounds__(256)
void gemm_wave(const _Float16* __restrict__ A, const _Float16* __restrict__ Bt,
               const float* __restrict__ bias, const float* __restrict__ res,
               _Float16* __restrict__ outh, float* __restrict__ outf,
               int M, int N, int K, int mode) {
    __shared__ __align__(16) _Float16 Bs[2][64 * KCHUNK];   // 2 x 16 KB

    const int wave   = threadIdx.x >> 5;
    const int lane   = threadIdx.x & 31;
    const int l16    = lane & 15;
    const int laneHi = lane >> 4;
    const int Nt     = N >> 6;
    const int tileN  = blockIdx.x % Nt;
    const int tileM  = (blockIdx.x / Nt) * 8 + wave;

    const _Float16* Arow  = A + (long)(tileM * 16 + l16) * K + laneHi * 8;
    const _Float16* Bg    = Bt + (long)tileN * 64 * K;   // 64 rows, stride K

    v8f acc[4] = {};
    const int nch = K / KCHUNK;

    // prologue: stage chunk 0
    if (wave == 0) {
        tdm_load_tile(Bg, (unsigned)(unsigned long long)(void*)&Bs[0][0],
                      64, KCHUNK, K);
        __builtin_amdgcn_s_wait_tensorcnt(0);
    }
    __syncthreads();

    for (int c = 0; c < nch; ++c) {
        // kick off DMA for next chunk while computing on this one
        if (wave == 0 && c + 1 < nch) {
            tdm_load_tile(Bg + (c + 1) * KCHUNK,
                          (unsigned)(unsigned long long)(void*)&Bs[(c + 1) & 1][0],
                          64, KCHUNK, K);
        }
        const _Float16* Bsc = &Bs[c & 1][0] + (long)l16 * KCHUNK + laneHi * 8;

        // A fragments for the whole chunk (independent global loads, issued early)
        v16h af[4];
        #pragma unroll
        for (int s = 0; s < 4; s++) {
            const _Float16* ap = Arow + c * KCHUNK + s * 32;
            ((v8h*)&af[s])[0] = *(const v8h*)ap;
            ((v8h*)&af[s])[1] = *(const v8h*)(ap + 16);
        }

        // B fragment loader for K-step s into a 4-fragment register set
        auto loadB = [&](v16h* bf, int s) {
            #pragma unroll
            for (int j = 0; j < 4; j++) {
                const _Float16* bp = Bsc + j * 16 * KCHUNK + s * 32;
                ((v8h*)&bf[j])[0] = *(const v8h*)bp;
                ((v8h*)&bf[j])[1] = *(const v8h*)(bp + 16);
            }
        };
        auto mma4 = [&](const v16h& a, const v16h* bf) {
            #pragma unroll
            for (int j = 0; j < 4; j++)
                acc[j] = __builtin_amdgcn_wmma_f32_16x16x32_f16(
                    false, a, false, bf[j], (short)0, acc[j], false, false);
        };

        // ping-pong: load step s+1 before computing step s
        v16h bfA[4], bfB[4];
        loadB(bfA, 0);
        loadB(bfB, 1);
        mma4(af[0], bfA);
        loadB(bfA, 2);
        mma4(af[1], bfB);
        loadB(bfB, 3);
        mma4(af[2], bfA);
        mma4(af[3], bfB);

        // pin the software pipeline: keep two B fragment sets in flight so
        // each DS-read group retires a full WMMA stage before it is consumed.
        // masks: 0x020 VMEM read, 0x100 DS read, 0x008 matrix ops
        __builtin_amdgcn_sched_group_barrier(0x020, 8, 0);   // 8 A-frag loads
        __builtin_amdgcn_sched_group_barrier(0x100, 16, 0);  // B frags s0+s1
        __builtin_amdgcn_sched_group_barrier(0x008, 4, 0);   // WMMA s0
        __builtin_amdgcn_sched_group_barrier(0x100, 8, 0);   // B frags s2
        __builtin_amdgcn_sched_group_barrier(0x008, 4, 0);   // WMMA s1
        __builtin_amdgcn_sched_group_barrier(0x100, 8, 0);   // B frags s3
        __builtin_amdgcn_sched_group_barrier(0x008, 8, 0);   // WMMA s2+s3

        // hand off: next buffer must be resident before anyone proceeds
        if (wave == 0 && c + 1 < nch)
            __builtin_amdgcn_s_wait_tensorcnt(0);
        __syncthreads();
    }

    #pragma unroll
    for (int j = 0; j < 4; j++) {
        int n = tileN * 64 + j * 16 + l16;
        float bv = bias[n];
        #pragma unroll
        for (int r = 0; r < 8; r++) {
            int m = tileM * 16 + r + laneHi * 8;
            float v = acc[j][r] + bv;
            if (mode == 1) v = v > 0.f ? v : 0.f;
            if (mode == 4) {
                outf[(long)m * N + n] = v + res[(long)m * N + n];
            } else if (mode == 2) {      // [B,H,S,Dh]
                int b_ = m >> 12, s_ = m & (SEQ - 1);
                int h_ = n >> 6,  dh = n & 63;
                outh[((((long)b_ * NHEADS + h_) * SEQ + s_) << 6) + dh] = (_Float16)v;
            } else if (mode == 3) {      // [B,H,Dh,S]
                int b_ = m >> 12, s_ = m & (SEQ - 1);
                int h_ = n >> 6,  dh = n & 63;
                outh[((((long)b_ * NHEADS + h_) * DHEAD + dh) << 12) + s_] = (_Float16)v;
            } else {
                outh[(long)m * N + n] = (_Float16)v;
            }
        }
    }
}

// ---------------------------------------------------------------------------
// Flash attention: 4 waves/block, each wave a 16-row Q tile for one (b,h).
// Q,K: [B,H,S,Dh] f16;  Vt: [B,H,Dh,S] f16;  ctx out: [M, D] f16
// ---------------------------------------------------------------------------
__global__ __launch_bounds__(128)
void attn_kernel(const _Float16* __restrict__ Q, const _Float16* __restrict__ Kb,
                 const _Float16* __restrict__ Vt, _Float16* __restrict__ ctx) {
    __shared__ __align__(16) _Float16 Pb[4][16 * 32];
    const int wid    = threadIdx.x >> 5;
    const int lane   = threadIdx.x & 31;
    const int l16    = lane & 15;
    const int laneHi = lane >> 4;
    int gw = blockIdx.x * 4 + wid;
    int qt = gw & (SEQ / 16 - 1);        // 0..255
    int bh = gw >> 8;                    // 0..15
    const float scale = 0.125f;          // 1/sqrt(64)

    const _Float16* Qp = Q + ((long)bh * SEQ + qt * 16 + l16) * DHEAD;
    v16h qa[2];
    #pragma unroll
    for (int ks = 0; ks < 2; ks++) {
        const _Float16* p = Qp + ks * 32 + laneHi * 8;
        ((v8h*)&qa[ks])[0] = *(const v8h*)p;
        ((v8h*)&qa[ks])[1] = *(const v8h*)(p + 16);
    }

    float mrow[8], lrow[8];
    #pragma unroll
    for (int r = 0; r < 8; r++) { mrow[r] = -1e30f; lrow[r] = 0.f; }
    v8f o[4] = {};

    const _Float16* Kbase = Kb + (long)bh * SEQ * DHEAD
                               + (long)l16 * DHEAD + laneHi * 8;
    const _Float16* Vbase = Vt + (long)bh * DHEAD * SEQ
                               + (long)l16 * SEQ + laneHi * 8;
    _Float16* Pl = &Pb[wid][0];

    for (int kb = 0; kb < SEQ; kb += 32) {
        // ---- load ALL fragments for this key block up front
        v16h kf[4];   // kf[ks*2+t]: K^T B-frag, key subtile t, dh half ks
        #pragma unroll
        for (int t = 0; t < 2; t++) {
            #pragma unroll
            for (int ks = 0; ks < 2; ks++) {
                const _Float16* bp = Kbase + (long)(kb + t * 16) * DHEAD + ks * 32;
                ((v8h*)&kf[ks*2+t])[0] = *(const v8h*)bp;
                ((v8h*)&kf[ks*2+t])[1] = *(const v8h*)(bp + 16);
            }
        }
        v16h vf[4];   // V B-frags: dh tile j, 32 keys contraction
        #pragma unroll
        for (int j = 0; j < 4; j++) {
            const _Float16* vp = Vbase + (long)j * 16 * SEQ + kb;
            ((v8h*)&vf[j])[0] = *(const v8h*)vp;
            ((v8h*)&vf[j])[1] = *(const v8h*)(vp + 16);
        }

        // ---- scores
        v8f s0 = {}, s1 = {};
        #pragma unroll
        for (int ks = 0; ks < 2; ks++) {
            s0 = __builtin_amdgcn_wmma_f32_16x16x32_f16(
                false, qa[ks], false, kf[ks*2+0], (short)0, s0, false, false);
            s1 = __builtin_amdgcn_wmma_f32_16x16x32_f16(
                false, qa[ks], false, kf[ks*2+1], (short)0, s1, false, false);
        }

        // ---- online softmax
        float p0[8], p1[8];
        #pragma unroll
        for (int r = 0; r < 8; r++) {
            float v0 = s0[r] * scale, v1 = s1[r] * scale;
            float rm = fmaxf(v0, v1);
            #pragma unroll
            for (int off = 1; off < 16; off <<= 1)
                rm = fmaxf(rm, __shfl_xor(rm, off, 16));
            float mn = fmaxf(mrow[r], rm);
            float e0 = __expf(v0 - mn), e1 = __expf(v1 - mn);
            float rs = e0 + e1;
            #pragma unroll
            for (int off = 1; off < 16; off <<= 1)
                rs += __shfl_xor(rs, off, 16);
            float alpha = __expf(mrow[r] - mn);
            lrow[r] = lrow[r] * alpha + rs;
            mrow[r] = mn;
            #pragma unroll
            for (int j = 0; j < 4; j++) o[j][r] *= alpha;
            p0[r] = e0; p1[r] = e1;
        }

        // ---- transpose P through per-wave LDS
        __syncthreads();
        #pragma unroll
        for (int r = 0; r < 8; r++) {
            int row = r + laneHi * 8;
            Pl[row * 32 + l16]      = (_Float16)p0[r];
            Pl[row * 32 + 16 + l16] = (_Float16)p1[r];
        }
        __syncthreads();
        v16h pa;
        const _Float16* pp = Pl + l16 * 32 + laneHi * 8;
        ((v8h*)&pa)[0] = *(const v8h*)pp;
        ((v8h*)&pa)[1] = *(const v8h*)(pp + 16);

        // ---- O += P @ V
        #pragma unroll
        for (int j = 0; j < 4; j++)
            o[j] = __builtin_amdgcn_wmma_f32_16x16x32_f16(
                false, pa, false, vf[j], (short)0, o[j], false, false);
    }

    // ---- normalize and write ctx as [M, D] f16
    int b_ = bh >> 3, h_ = bh & 7;
    #pragma unroll
    for (int r = 0; r < 8; r++) {
        float inv = 1.0f / lrow[r];
        int m = b_ * SEQ + qt * 16 + r + laneHi * 8;
        #pragma unroll
        for (int j = 0; j < 4; j++) {
            int col = h_ * DHEAD + j * 16 + l16;
            ctx[(long)m * DMODEL + col] = (_Float16)(o[j][r] * inv);
        }
    }
}

// ---------------------------------------------------------------------------
// host launcher
// ---------------------------------------------------------------------------
extern "C" void kernel_launch(void* const* d_in, const int* in_sizes, int n_in,
                              void* d_out, int out_size, void* d_ws, size_t ws_size,
                              hipStream_t stream) {
    const float* x    = (const float*)d_in[0];
    const float* Wq   = (const float*)d_in[1];
    const float* bq   = (const float*)d_in[2];
    const float* Wk   = (const float*)d_in[3];
    const float* bk   = (const float*)d_in[4];
    const float* Wv   = (const float*)d_in[5];
    const float* bv   = (const float*)d_in[6];
    const float* Wo   = (const float*)d_in[7];
    const float* bo   = (const float*)d_in[8];
    const float* ln1g = (const float*)d_in[9];
    const float* ln1b = (const float*)d_in[10];
    const float* W1   = (const float*)d_in[11];
    const float* b1   = (const float*)d_in[12];
    const float* W2   = (const float*)d_in[13];
    const float* b2   = (const float*)d_in[14];
    const float* ln2g = (const float*)d_in[15];
    const float* ln2b = (const float*)d_in[16];
    float* out = (float*)d_out;

    char* p = (char*)d_ws;
    auto take = [&](size_t bytes) {
        void* r = (void*)p;
        p += (bytes + 255) & ~(size_t)255;
        return r;
    };
    _Float16* WqT = (_Float16*)take((size_t)DMODEL * DMODEL * 2);
    _Float16* WkT = (_Float16*)take((size_t)DMODEL * DMODEL * 2);
    _Float16* WvT = (_Float16*)take((size_t)DMODEL * DMODEL * 2);
    _Float16* WoT = (_Float16*)take((size_t)DMODEL * DMODEL * 2);
    _Float16* W1T = (_Float16*)take((size_t)DMODEL * DFF * 2);
    _Float16* W2T = (_Float16*)take((size_t)DFF * DMODEL * 2);
    _Float16* xn  = (_Float16*)take((size_t)MROWS * DMODEL * 2);  // reused as ctx
    _Float16* Qb  = (_Float16*)take((size_t)MROWS * DMODEL * 2);  // reused as xn2
    _Float16* Kbf = (_Float16*)take((size_t)MROWS * DMODEL * 2);
    _Float16* Vt  = (_Float16*)take((size_t)MROWS * DMODEL * 2);
    _Float16* h1  = (_Float16*)take((size_t)MROWS * DFF * 2);
    float*    x1  = (float*)   take((size_t)MROWS * DMODEL * 4);
    _Float16* ctx = xn;
    _Float16* xn2 = Qb;

    // weight conversion + transpose
    convT_kernel<<<(DMODEL*DMODEL+255)/256, 256, 0, stream>>>(Wq, WqT, DMODEL, DMODEL);
    convT_kernel<<<(DMODEL*DMODEL+255)/256, 256, 0, stream>>>(Wk, WkT, DMODEL, DMODEL);
    convT_kernel<<<(DMODEL*DMODEL+255)/256, 256, 0, stream>>>(Wv, WvT, DMODEL, DMODEL);
    convT_kernel<<<(DMODEL*DMODEL+255)/256, 256, 0, stream>>>(Wo, WoT, DMODEL, DMODEL);
    convT_kernel<<<(DMODEL*DFF+255)/256,    256, 0, stream>>>(W1, W1T, DMODEL, DFF);
    convT_kernel<<<(DFF*DMODEL+255)/256,    256, 0, stream>>>(W2, W2T, DFF, DMODEL);

    // LN1
    ln_kernel<<<MROWS/8, 256, 0, stream>>>(x, ln1g, ln1b, xn);

    // QKV projections (scatter into attention-friendly layouts)
    // blocks = (Mt/8) * Nt
    int blkD = ((MROWS/16) / 8) * (DMODEL/64);   // 64 * 8  = 512
    gemm_wave<<<blkD, 256, 0, stream>>>(xn, WqT, bq, nullptr, Qb,  nullptr,
                                        MROWS, DMODEL, DMODEL, 2);
    gemm_wave<<<blkD, 256, 0, stream>>>(xn, WkT, bk, nullptr, Kbf, nullptr,
                                        MROWS, DMODEL, DMODEL, 2);
    gemm_wave<<<blkD, 256, 0, stream>>>(xn, WvT, bv, nullptr, Vt,  nullptr,
                                        MROWS, DMODEL, DMODEL, 3);

    // flash attention
    attn_kernel<<<BATCH*NHEADS*(SEQ/16)/4, 128, 0, stream>>>(Qb, Kbf, Vt, ctx);

    // output projection + residual -> x1 (fp32)
    gemm_wave<<<blkD, 256, 0, stream>>>(ctx, WoT, bo, x, nullptr, x1,
                                        MROWS, DMODEL, DMODEL, 4);

    // LN2
    ln_kernel<<<MROWS/8, 256, 0, stream>>>(x1, ln2g, ln2b, xn2);

    // FFN
    int blkF = ((MROWS/16) / 8) * (DFF/64);      // 64 * 32 = 2048
    gemm_wave<<<blkF, 256, 0, stream>>>(xn2, W1T, b1, nullptr, h1, nullptr,
                                        MROWS, DFF, DMODEL, 1);
    gemm_wave<<<blkD, 256, 0, stream>>>(h1, W2T, b2, x1, nullptr, out,
                                        MROWS, DMODEL, DFF, 4);
}
